// GraphContextForecaster_64458869178468
// MI455X (gfx1250) — compile-verified, matrix operand order checked by reading, not run on hardware
//
#include <hip/hip_runtime.h>
#include <hip/hip_bf16.h>
#include <math.h>

// MI455X / gfx1250: wave32, WMMA 16x16x32 bf16 for all 128-wide GEMMs.
// State kept fp32 (bandwidth-bound, x_m ~164MB ~= L2-resident), weights
// pre-packed to bf16 in exact WMMA B-matrix lane layout. Bulk GEMM stages
// the weight pack into LDS via the Tensor Data Mover (tensor_load_to_lds,
// TENSORcnt) and reads B tiles from LDS.

typedef __attribute__((ext_vector_type(16))) __bf16 v16bf;
typedef __attribute__((ext_vector_type(8)))  float  v8f;
typedef __attribute__((ext_vector_type(4)))  unsigned int u32x4;
typedef __attribute__((ext_vector_type(8)))  unsigned int u32x8;

#define NN 20000
#define TT 8
#define KK 4000
#define EE 400000
#define HD 128
#define MM 16

__device__ __forceinline__ float gelu_exact(float v) {
  return 0.5f * v * (1.0f + erff(v * 0.70710678118654752f));
}

// 16-bit A-matrix 16x32 layout: element-pair p (vector elems 2p,2p+1) of lane
// (m = lane&15, hi = lane>>4) holds K = koff(p,hi), koff(p,hi)+1 within a chunk.
__device__ __forceinline__ int a_koff(int p, int hi) {
  return (p < 4) ? (2 * p + 8 * hi) : (16 + 2 * (p - 4) + 8 * hi);
}

// TDM: DMA one packed weight matrix (16384 bf16 = 32KB, contiguous) into LDS.
// 2D descriptor, 2-SGPR-group form (<=2D tensors): D# group0 + group1.
__device__ __forceinline__ void tdm_load_weights(unsigned int lds_off,
                                                 const void* gptr) {
  unsigned long long ga = (unsigned long long)(uintptr_t)gptr;
  u32x4 g0;
  g0[0] = 1u;                                        // count=1, user descriptor
  g0[1] = lds_off;                                   // lds_addr (bytes)
  g0[2] = (unsigned int)(ga & 0xffffffffu);          // global_addr[31:0]
  g0[3] = (unsigned int)((ga >> 32) & 0x01ffffffu)   // global_addr[56:32]
          | (2u << 30);                              // type = 2 ("image")
  u32x8 g1;
  g1[0] = 1u << 16;                                  // data_size=1 -> 2 bytes
  g1[1] = (16384u & 0xffffu) << 16;                  // tensor_dim0[15:0]
  g1[2] = (16384u >> 16) | (1u << 16);               // dim0[31:16] | tensor_dim1=1
  g1[3] = 16384u << 16;                              // tile_dim0 = 16384
  g1[4] = 1u;                                        // tile_dim1 = 1
  g1[5] = 16384u;                                    // tensor_dim0_stride[31:0]
  g1[6] = 0u;
  g1[7] = 0u;
  asm volatile("tensor_load_to_lds %0, %1" :: "s"(g0), "s"(g1) : "memory");
  __builtin_amdgcn_s_wait_tensorcnt(0);
}

// ---------------------------------------------------------------- utilities
__global__ void k_zero(float* __restrict__ p, long long n) {
  long long i = (long long)blockIdx.x * blockDim.x + threadIdx.x;
  if (i < n) p[i] = 0.f;
}

// Pack W [out=128][in=128] f32 row-major into bf16 B-matrix tiles:
// B[k][n] = W[n][k]; flat = (((tn*4+kc)*32 + lane)*8 + p)*2 + half,
// k = kc*32 + koff(p, lane>>4) + half, n = tn*16 + (lane&15).
__global__ void k_packW(const float* __restrict__ W, __bf16* __restrict__ P) {
  int idx = blockIdx.x * 256 + threadIdx.x;            // 16384 total
  int q  = idx & 15;
  int L  = (idx >> 4) & 31;
  int kc = (idx >> 9) & 3;
  int tn = idx >> 11;
  int p = q >> 1, half = q & 1, hi = L >> 4;
  int k = kc * 32 + a_koff(p, hi) + half;
  int n = tn * 16 + (L & 15);
  P[idx] = (__bf16)W[n * HD + k];
}

// ---------------------------------------------------------------- GEMM 128
// out[R,128] = act(in[R,128]) @ (packed W^T), one 16-row tile per wave.
// Weight pack staged to LDS via TDM once per workgroup.
__global__ void k_gemm128(const float* __restrict__ in,
                          const __bf16* __restrict__ Wp,
                          float* __restrict__ out, int R, int doRelu) {
  __shared__ __bf16 Wl[16384];                 // 32 KB staged weights
  int lane = threadIdx.x & 31;
  int wave = threadIdx.x >> 5;

  if (threadIdx.x < 32)                        // wave 0 issues the DMA
    tdm_load_weights((unsigned int)(uintptr_t)&Wl[0], Wp);
  __syncthreads();

  int tile = blockIdx.x * 8 + wave;
  int ntiles = R >> 4;
  if (tile >= ntiles) return;                  // wave-uniform exit (EXEC all-1 for WMMA)
  int m  = tile * 16 + (lane & 15);
  int hi = lane >> 4;
  const float* arow = in + (size_t)m * HD;

  v16bf a4[4];
#pragma unroll
  for (int kc = 0; kc < 4; kc++) {
#pragma unroll
    for (int p = 0; p < 8; p++) {
      int kb = kc * 32 + a_koff(p, hi);
      float v0 = arow[kb], v1 = arow[kb + 1];
      if (doRelu) { v0 = fmaxf(v0, 0.f); v1 = fmaxf(v1, 0.f); }
      a4[kc][2 * p]     = (__bf16)v0;
      a4[kc][2 * p + 1] = (__bf16)v1;
    }
  }
  float* orow = out + (size_t)tile * 16 * HD;
#pragma unroll
  for (int tn = 0; tn < 8; tn++) {
    v8f acc = {};
#pragma unroll
    for (int kc = 0; kc < 4; kc++) {
      v16bf b = *reinterpret_cast<const v16bf*>(&Wl[((tn * 4 + kc) * 32 + lane) * 16]);
      acc = __builtin_amdgcn_wmma_f32_16x16x32_bf16(false, a4[kc], false, b,
                                                    (short)0, acc, false, false);
    }
    int col = tn * 16 + (lane & 15);
#pragma unroll
    for (int r = 0; r < 8; r++)
      orow[(size_t)(r + 8 * hi) * HD + col] = acc[r];
  }
}

// ---------------------------------------------------------------- token mix
// xt[n][j][h] (logical order) = x[n][j][h] + (tokMLP(LN(x))^T)[j][h]
__global__ void k_token_mix(const float* __restrict__ X, float* __restrict__ xt,
                            const float* __restrict__ tg, const float* __restrict__ tb,
                            const float* __restrict__ w1, const float* __restrict__ b1,
                            const float* __restrict__ w2, const float* __restrict__ b2,
                            int origin) {
  __shared__ float sw1[256], sw2[256], sb1[16], sb2[16], smu[16], srs[16];
  int n = blockIdx.x;
  int h = threadIdx.x;                       // 128 threads
  const float* x = X + (size_t)n * MM * HD;

  sw1[h] = w1[h];  sw1[h + 128] = w1[h + 128];
  sw2[h] = w2[h];  sw2[h + 128] = w2[h + 128];
  if (h < 16) { sb1[h] = b1[h]; sb2[h] = b2[h]; }
  if (h < 16) {
    int pr = (origin + h) & 15;
    float s = 0.f, s2 = 0.f;
    for (int k = 0; k < HD; k++) { float v = x[pr * HD + k]; s += v; s2 += v * v; }
    float mu = s * (1.f / HD);
    float var = s2 * (1.f / HD) - mu * mu;
    smu[h] = mu;
    srs[h] = rsqrtf(var + 1e-5f);
  }
  __syncthreads();

  float g = tg[h], bb = tb[h];
  float v[16], u[16];
#pragma unroll
  for (int j = 0; j < 16; j++) {
    int pr = (origin + j) & 15;
    v[j] = (x[pr * HD + h] - smu[j]) * srs[j] * g + bb;
  }
#pragma unroll
  for (int i = 0; i < 16; i++) {
    float a = sb1[i];
#pragma unroll
    for (int j = 0; j < 16; j++) a += v[j] * sw1[i * 16 + j];
    u[i] = gelu_exact(a);
  }
#pragma unroll
  for (int i = 0; i < 16; i++) {
    float a = sb2[i];
#pragma unroll
    for (int j = 0; j < 16; j++) a += u[j] * sw2[i * 16 + j];
    int pr = (origin + i) & 15;
    xt[(size_t)n * MM * HD + i * HD + h] = x[pr * HD + h] + a;
  }
}

// ------------------------------------------------------------- channel mix
// One wave per node: out = xt + chMLP(LN(xt)); written back to X in physical
// slot order; also writes step output = mean over the 16 slots.
__global__ void k_channel_mix(const float* __restrict__ xt, float* __restrict__ X,
                              float* __restrict__ out,
                              const float* __restrict__ cg, const float* __restrict__ cb,
                              const __bf16* __restrict__ W1p, const float* __restrict__ b1,
                              const __bf16* __restrict__ W2p, const float* __restrict__ b2,
                              int origin) {
  __shared__ __bf16 h1s[8][MM][HD];          // 32 KB, per-wave staging of gelu(layer1)
  int lane = threadIdx.x & 31;
  int wave = threadIdx.x >> 5;
  int node = blockIdx.x * 8 + wave;
  const float* x = xt + (size_t)node * MM * HD;
  int mrow = lane & 15;
  int hi   = lane >> 4;

  // LayerNorm stats for row mrow (duplicated across lane halves; L1-resident)
  float s = 0.f, s2 = 0.f;
  for (int k = 0; k < HD; k++) { float v = x[mrow * HD + k]; s += v; s2 += v * v; }
  float mu = s * (1.f / HD);
  float rs = rsqrtf(s2 * (1.f / HD) - mu * mu + 1e-5f);

  // A (layer 1) = LN(xt) in WMMA A layout
  v16bf a4[4];
#pragma unroll
  for (int kc = 0; kc < 4; kc++) {
#pragma unroll
    for (int p = 0; p < 8; p++) {
      int kb = kc * 32 + a_koff(p, hi);
      a4[kc][2 * p]     = (__bf16)((x[mrow * HD + kb]     - mu) * rs * cg[kb]     + cb[kb]);
      a4[kc][2 * p + 1] = (__bf16)((x[mrow * HD + kb + 1] - mu) * rs * cg[kb + 1] + cb[kb + 1]);
    }
  }
#pragma unroll
  for (int tn = 0; tn < 8; tn++) {
    v8f acc = {};
#pragma unroll
    for (int kc = 0; kc < 4; kc++) {
      v16bf b = *reinterpret_cast<const v16bf*>(W1p + ((size_t)(tn * 4 + kc) * 32 + lane) * 16);
      acc = __builtin_amdgcn_wmma_f32_16x16x32_bf16(false, a4[kc], false, b,
                                                    (short)0, acc, false, false);
    }
    int col = tn * 16 + mrow;
    float bias = b1[col];
#pragma unroll
    for (int r = 0; r < 8; r++)
      h1s[wave][r + 8 * hi][col] = (__bf16)gelu_exact(acc[r] + bias);
  }
  __syncthreads();

  // A (layer 2) = gelu activations from LDS
  v16bf c4[4];
#pragma unroll
  for (int kc = 0; kc < 4; kc++) {
#pragma unroll
    for (int p = 0; p < 8; p++) {
      int kb = kc * 32 + a_koff(p, hi);
      c4[kc][2 * p]     = h1s[wave][mrow][kb];
      c4[kc][2 * p + 1] = h1s[wave][mrow][kb + 1];
    }
  }
  float* Xn = X + (size_t)node * MM * HD;
#pragma unroll
  for (int tn = 0; tn < 8; tn++) {
    v8f acc = {};
#pragma unroll
    for (int kc = 0; kc < 4; kc++) {
      v16bf b = *reinterpret_cast<const v16bf*>(W2p + ((size_t)(tn * 4 + kc) * 32 + lane) * 16);
      acc = __builtin_amdgcn_wmma_f32_16x16x32_bf16(false, c4[kc], false, b,
                                                    (short)0, acc, false, false);
    }
    int col = tn * 16 + mrow;
    float bias = b2[col];
    float csum = 0.f;
#pragma unroll
    for (int r = 0; r < 8; r++) {
      int lrow = r + 8 * hi;                         // logical slot
      float v = x[lrow * HD + col] + acc[r] + bias;  // residual
      csum += v;
      Xn[((origin + lrow) & 15) * HD + col] = v;     // write physical slot
    }
    csum += __shfl_xor(csum, 16);
    if (hi == 0) out[(size_t)node * HD + col] = csum * (1.f / MM);
  }
}

// ------------------------------------------------------------- GCN pieces
__global__ void k_deg_init(float* __restrict__ deg) {
  int n = blockIdx.x * 256 + threadIdx.x;
  if (n < NN) deg[n] = 1.f;                          // self-loop
}
__global__ void k_deg_acc(const int* __restrict__ dst, float* __restrict__ deg) {
  int e = blockIdx.x * 256 + threadIdx.x;
  if (e < EE) unsafeAtomicAdd(&deg[dst[e]], 1.f);
}
// out[n][h] = bias[h] + dinv[n]^2 * in[n][h]   (self-loop term)
__global__ void k_agg_init(const float* __restrict__ in, const float* __restrict__ deg,
                           const float* __restrict__ bias, float* __restrict__ out) {
  int idx = blockIdx.x * 256 + threadIdx.x;          // N*128 exact
  int n = idx >> 7, h = idx & 127;
  float di = rsqrtf(fmaxf(deg[n], 1.f));
  out[idx] = bias[h] + di * di * in[idx];
}
__global__ void k_agg_edges(const int* __restrict__ src, const int* __restrict__ dst,
                            const float* __restrict__ in, const float* __restrict__ deg,
                            float* __restrict__ out) {
  int idx = blockIdx.x * 256 + threadIdx.x;          // E*128 exact
  int e = idx >> 7, h = idx & 127;
  int s = src[e], d = dst[e];
  float nrm = rsqrtf(fmaxf(deg[s], 1.f)) * rsqrtf(fmaxf(deg[d], 1.f));
  unsafeAtomicAdd(&out[(size_t)d * HD + h], in[(size_t)s * HD + h] * nrm);
}
__global__ void k_scatter_zero(const int* __restrict__ nodes, float* __restrict__ A) {
  int n = nodes[blockIdx.x];
  A[(size_t)n * HD + threadIdx.x] = 0.f;
}
__global__ void k_insert(const float* __restrict__ A, float* __restrict__ X, int slot) {
  int idx = blockIdx.x * 256 + threadIdx.x;          // N*128 exact
  int n = idx >> 7, h = idx & 127;
  X[((size_t)n * MM + slot) * HD + h] = A[idx];
}

// ---------------------------------------------------------------- launcher
extern "C" void kernel_launch(void* const* d_in, const int* in_sizes, int n_in,
                              void* d_out, int out_size, void* d_ws, size_t ws_size,
                              hipStream_t stream) {
  (void)in_sizes; (void)n_in; (void)out_size; (void)ws_size;
  const int*   node_t = (const int*)d_in[0];
  const int*   edge_t = (const int*)d_in[1];
  const float* emb    = (const float*)d_in[3];
  const float* convW1 = (const float*)d_in[4];
  const float* convb1 = (const float*)d_in[5];
  const float* convW2 = (const float*)d_in[6];
  const float* convb2 = (const float*)d_in[7];
  const float* tn_g   = (const float*)d_in[8];
  const float* tn_b   = (const float*)d_in[9];
  const float* tok_w1 = (const float*)d_in[10];
  const float* tok_b1 = (const float*)d_in[11];
  const float* tok_w2 = (const float*)d_in[12];
  const float* tok_b2 = (const float*)d_in[13];
  const float* cn_g   = (const float*)d_in[14];
  const float* cn_b   = (const float*)d_in[15];
  const float* ch_w1  = (const float*)d_in[16];
  const float* ch_b1  = (const float*)d_in[17];
  const float* ch_w2  = (const float*)d_in[18];
  const float* ch_b2  = (const float*)d_in[19];
  float* out = (float*)d_out;

  char* wp = (char*)d_ws;
  auto carve = [&](size_t bytes) -> void* {
    void* p = (void*)wp;
    wp += (bytes + 255) & ~(size_t)255;
    return p;
  };
  const size_t stateElems = (size_t)NN * MM * HD;
  float*  X   = (float*)carve(stateElems * 4);       // circular window state
  float*  XT  = (float*)carve(stateElems * 4);       // token-mix output (logical order)
  float*  XW1 = (float*)carve((size_t)NN * HD * 4);  // emb @ convW1^T
  float*  A1  = (float*)carve((size_t)NN * HD * 4);
  float*  A2  = (float*)carve((size_t)NN * HD * 4);
  float*  A3  = (float*)carve((size_t)NN * HD * 4);
  float*  DEG = (float*)carve((size_t)NN * 4);
  __bf16* PC1 = (__bf16*)carve(16384 * 2);           // convW1 pack
  __bf16* PC2 = (__bf16*)carve(16384 * 2);           // convW2 pack
  __bf16* PH1 = (__bf16*)carve(16384 * 2);           // ch_w1 pack
  __bf16* PH2 = (__bf16*)carve(16384 * 2);           // ch_w2 pack

  // ---- one-time setup
  k_zero<<<(int)((stateElems + 255) / 256), 256, 0, stream>>>(X, (long long)stateElems);
  k_packW<<<64, 256, 0, stream>>>(convW1, PC1);
  k_packW<<<64, 256, 0, stream>>>(convW2, PC2);
  k_packW<<<64, 256, 0, stream>>>(ch_w1, PH1);
  k_packW<<<64, 256, 0, stream>>>(ch_w2, PH2);
  // xw1 = emb @ convW1^T : 1250 tiles, 8 waves/block
  k_gemm128<<<157, 256, 0, stream>>>(emb, PC1, XW1, NN, 0);

  // ---- temporal scan
  for (int t = 0; t < TT; t++) {
    const int* src = edge_t + (size_t)t * 2 * EE;
    const int* dst = src + EE;

    // mixer
    k_token_mix<<<NN, 128, 0, stream>>>(X, XT, tn_g, tn_b, tok_w1, tok_b1,
                                        tok_w2, tok_b2, t);
    k_channel_mix<<<NN / 8, 256, 0, stream>>>(XT, X, out + (size_t)t * NN * HD,
                                              cn_g, cn_b, PH1, ch_b1, PH2, ch_b2, t);
    // degrees (with self-loops)
    k_deg_init<<<(NN + 255) / 256, 256, 0, stream>>>(DEG);
    k_deg_acc<<<(EE + 255) / 256, 256, 0, stream>>>(dst, DEG);
    // GCN layer 1: A1 = segsum(norm * xw1[src]) + b1 ; relu fused into GEMM
    k_agg_init<<<NN * HD / 256, 256, 0, stream>>>(XW1, DEG, convb1, A1);
    k_agg_edges<<<(size_t)EE * HD / 256, 256, 0, stream>>>(src, dst, XW1, DEG, A1);
    // A2 = relu(A1) @ convW2^T
    k_gemm128<<<157, 256, 0, stream>>>(A1, PC2, A2, NN, 1);
    // GCN layer 2 aggregation
    k_agg_init<<<NN * HD / 256, 256, 0, stream>>>(A2, DEG, convb2, A3);
    k_agg_edges<<<(size_t)EE * HD / 256, 256, 0, stream>>>(src, dst, A2, DEG, A3);
    // zero rows of nodes present this step, then slide window (circular slot)
    k_scatter_zero<<<KK, 128, 0, stream>>>(node_t + (size_t)t * KK, A3);
    k_insert<<<NN * HD / 256, 256, 0, stream>>>(A3, X, t & 15);
  }
}
